// MultiHeadAttention_31842887532549
// MI455X (gfx1250) — matrix operand order repaired and checked by reading, not compiled, not run on hardware
//
#include <hip/hip_runtime.h>
#include <hip/hip_bf16.h>

typedef __attribute__((ext_vector_type(4)))  _Float16 v4h;
typedef __attribute__((ext_vector_type(8)))  _Float16 v8h;
typedef __attribute__((ext_vector_type(16))) _Float16 v16h;
typedef __attribute__((ext_vector_type(8)))  float    v8f;

#define BATCH  4
#define SEQ    2048
#define DMODEL 1024
#define NHEAD  16
#define DHEAD  64
#define MROWS  (BATCH * SEQ)   // 8192

__device__ __forceinline__ v8f wmma16(v16h a, v16h b, v8f c) {
  // v_wmma_f32_16x16x32_f16: D = A(16x32,f16) * B(32x16,f16) + C(f32)
  return __builtin_amdgcn_wmma_f32_16x16x32_f16(false, a, false, b, (short)0, c,
                                                false, false);
}

// A-fragment (16x32 f16): lane holds row M=lane&15; K = half*8 + e (e<8) and
// 16 + half*8 + (e-8) (e>=8). Caller passes p = &row[half*8]; halves at p, p+16.
__device__ __forceinline__ v16h load_fragA(const _Float16* p) {
  v8h lo = *(const v8h*)p;
  v8h hi = *(const v8h*)(p + 16);
  return __builtin_shufflevector(lo, hi, 0, 1, 2, 3, 4, 5, 6, 7,
                                 8, 9, 10, 11, 12, 13, 14, 15);
}

// B-fragment (32x16 f16): lane holds column N=lane&15; K = half*8 + e (e<8) and
// 16 + half*8 + (e-8) (e>=8). Caller passes p = &col[half*8]; halves at p, p+16.
__device__ __forceinline__ v16h load_fragB(const _Float16* p) {
  v8h lo = *(const v8h*)p;
  v8h hi = *(const v8h*)(p + 16);
  return __builtin_shufflevector(lo, hi, 0, 1, 2, 3, 4, 5, 6, 7,
                                 8, 9, 10, 11, 12, 13, 14, 15);
}

// CDNA5 async global->LDS copy (16B per lane), tracked on ASYNCcnt.
__device__ __forceinline__ void async_copy_b128(uint32_t lds_off,
                                                const _Float16* gptr) {
  asm volatile("global_load_async_to_lds_b128 %0, %1, off"
               :: "v"(lds_off), "v"(gptr)
               : "memory");
}
__device__ __forceinline__ void wait_asynccnt0() {
  asm volatile("s_wait_asynccnt 0x0" ::: "memory");
}

// ---------------------------------------------------------------------------
// Kernel 1: fused QKV projection. z selects q/k/v. Output f16:
//   Q,K -> [B,H,S,DH] (head rows contiguous); V -> transposed [B,H,DH,S].
// Block: 256 threads (8 waves), tile 128(M) x 128(N), K-step 32.
// Each wave computes 32x64 via 2x4 WMMA tiles (8 WMMAs / k-step).
// ---------------------------------------------------------------------------
__global__ __launch_bounds__(256) void qkv_proj_kernel(
    const float* __restrict__ qin, const float* __restrict__ kin,
    const float* __restrict__ vin,
    const float* __restrict__ Wq, const float* __restrict__ Wk,
    const float* __restrict__ Wv,
    const float* __restrict__ bq, const float* __restrict__ bk,
    const float* __restrict__ bv,
    _Float16* __restrict__ Qh, _Float16* __restrict__ Kh,
    _Float16* __restrict__ Vt) {
  __shared__ __align__(16) _Float16 As[128][40];  // [m][k], +8 halves pad
  __shared__ __align__(16) _Float16 Bs[128][40];  // [n][k] (transposed W tile)

  const int mode = blockIdx.z;
  const float* x    = (mode == 0) ? qin : (mode == 1) ? kin : vin;
  const float* W    = (mode == 0) ? Wq  : (mode == 1) ? Wk  : Wv;
  const float* bias = (mode == 0) ? bq  : (mode == 1) ? bk  : bv;

  const int mBase = blockIdx.x * 128;
  const int nBase = blockIdx.y * 128;
  const int t    = threadIdx.x;
  const int wave = t >> 5, lane = t & 31;
  const int half = lane >> 4, ln = lane & 15;
  const int wm = wave & 3, wn = wave >> 2;  // wave tile: rows wm*32, cols wn*64

  v8f acc[2][4] = {};

  for (int k0 = 0; k0 < DMODEL; k0 += 32) {
    // X tile 128x32 f32 -> f16 LDS (4 float4 per thread, packed b64 stores)
#pragma unroll
    for (int i = 0; i < 4; ++i) {
      int uu = t + i * 256;            // 0..1023
      int row = uu >> 3, c4 = uu & 7;  // 8 float4 per row
      float4 f = *(const float4*)(x + (size_t)(mBase + row) * DMODEL + k0 + c4 * 4);
      v4h hv;
      hv[0] = (_Float16)f.x; hv[1] = (_Float16)f.y;
      hv[2] = (_Float16)f.z; hv[3] = (_Float16)f.w;
      *(v4h*)&As[row][c4 * 4] = hv;
    }
    // W tile 32x128 f32 -> f16, transposed into Bs[n][k]
#pragma unroll
    for (int i = 0; i < 4; ++i) {
      int uu = t + i * 256;             // 0..1023
      int row = uu >> 5, c4 = uu & 31;  // 32 float4 per k-row
      float4 f = *(const float4*)(W + (size_t)(k0 + row) * DMODEL + nBase + c4 * 4);
      Bs[c4 * 4 + 0][row] = (_Float16)f.x;
      Bs[c4 * 4 + 1][row] = (_Float16)f.y;
      Bs[c4 * 4 + 2][row] = (_Float16)f.z;
      Bs[c4 * 4 + 3][row] = (_Float16)f.w;
    }
    __syncthreads();

    v16h af[2], bf[4];
#pragma unroll
    for (int mt = 0; mt < 2; ++mt)
      af[mt] = load_fragA(&As[wm * 32 + mt * 16 + ln][half * 8]);
#pragma unroll
    for (int nt = 0; nt < 4; ++nt)
      bf[nt] = load_fragB(&Bs[wn * 64 + nt * 16 + ln][half * 8]);
#pragma unroll
    for (int mt = 0; mt < 2; ++mt)
#pragma unroll
      for (int nt = 0; nt < 4; ++nt)
        acc[mt][nt] = wmma16(af[mt], bf[nt], acc[mt][nt]);
    __syncthreads();
  }

  // epilogue: + bias, convert f16, scatter to attention-friendly layouts
#pragma unroll
  for (int nt = 0; nt < 4; ++nt) {
    int nCol = nBase + wn * 64 + nt * 16 + ln;
    float bval = bias[nCol];
    int hh = nCol >> 6, dh = nCol & 63;
#pragma unroll
    for (int mt = 0; mt < 2; ++mt) {
#pragma unroll
      for (int r = 0; r < 8; ++r) {
        int g = mBase + wm * 32 + mt * 16 + half * 8 + r;  // global row = b*S+s
        int bidx = g >> 11, s = g & 2047;
        _Float16 hv = (_Float16)(acc[mt][nt][r] + bval);
        if (mode == 0)
          Qh[((size_t)(bidx * NHEAD + hh) * SEQ + s) * DHEAD + dh] = hv;
        else if (mode == 1)
          Kh[((size_t)(bidx * NHEAD + hh) * SEQ + s) * DHEAD + dh] = hv;
        else
          Vt[((size_t)(bidx * NHEAD + hh) * DHEAD + dh) * SEQ + s] = hv;
      }
    }
  }
}

// ---------------------------------------------------------------------------
// Kernel 2: flash attention. Block = (q-tile of 64, head, batch), 128 threads
// (4 waves); each wave owns 16 query rows and streams all 2048 keys in tiles
// of 32, with online softmax. K/V tiles staged in LDS via CDNA5 async
// global->LDS copies, shared by all 4 waves.
// ---------------------------------------------------------------------------
__global__ __launch_bounds__(128) void flash_attn_kernel(
    const _Float16* __restrict__ Qh, const _Float16* __restrict__ Kh,
    const _Float16* __restrict__ Vt, _Float16* __restrict__ attn) {
  __shared__ __align__(16) _Float16 Ks[32][72];      // [key_local][dh]
  __shared__ __align__(16) _Float16 Vs[64][40];      // [d][key_local]
  __shared__ __align__(16) _Float16 Ps[4][16][40];   // per-wave P tile 16x32

  const int qt = blockIdx.x, h = blockIdx.y, b = blockIdx.z;
  const int t = threadIdx.x;
  const int wave = t >> 5, lane = t & 31;
  const int half = lane >> 4, ln = lane & 15;

  const _Float16* Qb = Qh + (size_t)(b * NHEAD + h) * SEQ * DHEAD;
  const _Float16* Kb = Kh + (size_t)(b * NHEAD + h) * SEQ * DHEAD;
  const _Float16* Vb = Vt + (size_t)(b * NHEAD + h) * DHEAD * SEQ;

  const int qBase = qt * 64 + wave * 16;

  // per-lane staging slots (2 K-chunks, 2 V-chunks per lane per tile)
  const int krow0 = t >> 3,          kc80 = t & 7;          // K: chunks 0..127
  const int krow1 = (t + 128) >> 3,  kc81 = (t + 128) & 7;  // K: chunks 128..255
  const int vrow0 = t >> 2,          vc80 = t & 3;          // V: chunks 0..127
  const int vrow1 = (t + 128) >> 2,  vc81 = (t + 128) & 3;  // V: chunks 128..255
  const uint32_t ldsK0 = (uint32_t)(uintptr_t)&Ks[krow0][kc80 * 8];
  const uint32_t ldsK1 = (uint32_t)(uintptr_t)&Ks[krow1][kc81 * 8];
  const uint32_t ldsV0 = (uint32_t)(uintptr_t)&Vs[vrow0][vc80 * 8];
  const uint32_t ldsV1 = (uint32_t)(uintptr_t)&Vs[vrow1][vc81 * 8];

  // preload Q A-fragments (16 rows x DH=64 -> 2 K-chunks of 32)
  v16h qf[2];
#pragma unroll
  for (int kc = 0; kc < 2; ++kc)
    qf[kc] = load_fragA(Qb + (size_t)(qBase + ln) * DHEAD + kc * 32 + half * 8);

  float mi[8], li[8];
  v8f o[4] = {};
#pragma unroll
  for (int r = 0; r < 8; ++r) { mi[r] = -1e30f; li[r] = 0.f; }

  for (int kt = 0; kt < SEQ / 32; ++kt) {
    // async stage: K tile 32x64, V tile 64x32 (f16, 16B per lane per op)
    async_copy_b128(ldsK0, Kb + (size_t)(kt * 32 + krow0) * DHEAD + kc80 * 8);
    async_copy_b128(ldsK1, Kb + (size_t)(kt * 32 + krow1) * DHEAD + kc81 * 8);
    async_copy_b128(ldsV0, Vb + (size_t)vrow0 * SEQ + kt * 32 + vc80 * 8);
    async_copy_b128(ldsV1, Vb + (size_t)vrow1 * SEQ + kt * 32 + vc81 * 8);
    wait_asynccnt0();
    __syncthreads();

    // S = Q * K^T * scale  (two 16-column tiles)
    v8f sf[2];
#pragma unroll
    for (int nt = 0; nt < 2; ++nt) {
      v8f accv = {};
#pragma unroll
      for (int kc = 0; kc < 2; ++kc) {
        v16h bf = load_fragB(&Ks[nt * 16 + ln][kc * 32 + half * 8]);
        accv = wmma16(qf[kc], bf, accv);
      }
      sf[nt] = accv * 0.125f;  // 1/sqrt(64)
    }

    // online softmax: C-layout row r lives in one 16-lane half-group
#pragma unroll
    for (int r = 0; r < 8; ++r) {
      float mx = fmaxf(sf[0][r], sf[1][r]);
      mx = fmaxf(mx, __shfl_xor(mx, 1));
      mx = fmaxf(mx, __shfl_xor(mx, 2));
      mx = fmaxf(mx, __shfl_xor(mx, 4));
      mx = fmaxf(mx, __shfl_xor(mx, 8));
      float mnew  = fmaxf(mi[r], mx);
      float alpha = __expf(mi[r] - mnew);
      float p0 = __expf(sf[0][r] - mnew);
      float p1 = __expf(sf[1][r] - mnew);
      sf[0][r] = p0; sf[1][r] = p1;
      float rs = p0 + p1;
      rs += __shfl_xor(rs, 1);
      rs += __shfl_xor(rs, 2);
      rs += __shfl_xor(rs, 4);
      rs += __shfl_xor(rs, 8);
      li[r] = li[r] * alpha + rs;
      mi[r] = mnew;
#pragma unroll
      for (int ot = 0; ot < 4; ++ot) o[ot][r] *= alpha;
    }

    // C-layout -> A-layout for P via per-wave LDS (same-wave DS is in-order)
#pragma unroll
    for (int r = 0; r < 8; ++r) {
      Ps[wave][half * 8 + r][ln]      = (_Float16)sf[0][r];
      Ps[wave][half * 8 + r][16 + ln] = (_Float16)sf[1][r];
    }
    v16h pf = load_fragA(&Ps[wave][ln][half * 8]);

    // O += P * V  (4 output column tiles of 16)
#pragma unroll
    for (int ot = 0; ot < 4; ++ot) {
      v16h bf = load_fragB(&Vs[ot * 16 + ln][half * 8]);
      o[ot] = wmma16(pf, bf, o[ot]);
    }
    __syncthreads();
  }

  // normalize and write attention output as [B,S,D] f16 (row-major for GEMM3)
#pragma unroll
  for (int ot = 0; ot < 4; ++ot) {
#pragma unroll
    for (int r = 0; r < 8; ++r) {
      float val = o[ot][r] / li[r];
      int srow = qBase + half * 8 + r;
      attn[((size_t)b * SEQ + srow) * DMODEL + h * DHEAD + ot * 16 + ln] =
          (_Float16)val;
    }
  }
}

// ---------------------------------------------------------------------------
// Kernel 3: output projection, f16 A x f16(Wo) + bo -> fp32 d_out.
// Tile 128x128, 8 waves, 8 WMMAs per wave per k-step.
// ---------------------------------------------------------------------------
__global__ __launch_bounds__(256) void out_proj_kernel(
    const _Float16* __restrict__ Ain, const float* __restrict__ Wo,
    const float* __restrict__ bo, float* __restrict__ out) {
  __shared__ __align__(16) _Float16 As[128][40];
  __shared__ __align__(16) _Float16 Bs[128][40];

  const int mBase = blockIdx.x * 128;
  const int nBase = blockIdx.y * 128;
  const int t = threadIdx.x;
  const int wave = t >> 5, lane = t & 31;
  const int half = lane >> 4, ln = lane & 15;
  const int wm = wave & 3, wn = wave >> 2;

  v8f acc[2][4] = {};

  for (int k0 = 0; k0 < DMODEL; k0 += 32) {
#pragma unroll
    for (int i = 0; i < 2; ++i) {
      int uu = t + i * 256;            // 0..511
      int row = uu >> 2, c8 = uu & 3;  // 4 v8h per row
      *(v8h*)&As[row][c8 * 8] =
          *(const v8h*)(Ain + (size_t)(mBase + row) * DMODEL + k0 + c8 * 8);
    }
#pragma unroll
    for (int i = 0; i < 4; ++i) {
      int uu = t + i * 256;             // 0..1023
      int row = uu >> 5, c4 = uu & 31;  // 32 float4 per k-row
      float4 f = *(const float4*)(Wo + (size_t)(k0 + row) * DMODEL + nBase + c4 * 4);
      Bs[c4 * 4 + 0][row] = (_Float16)f.x;
      Bs[c4 * 4 + 1][row] = (_Float16)f.y;
      Bs[c4 * 4 + 2][row] = (_Float16)f.z;
      Bs[c4 * 4 + 3][row] = (_Float16)f.w;
    }
    __syncthreads();

    v16h af[2], bf[4];
#pragma unroll
    for (int mt = 0; mt < 2; ++mt)
      af[mt] = load_fragA(&As[wm * 32 + mt * 16 + ln][half * 8]);
#pragma unroll
    for (int nt = 0; nt < 4; ++nt)
      bf[nt] = load_fragB(&Bs[wn * 64 + nt * 16 + ln][half * 8]);
#pragma unroll
    for (int mt = 0; mt < 2; ++mt)
#pragma unroll
      for (int nt = 0; nt < 4; ++nt)
        acc[mt][nt] = wmma16(af[mt], bf[nt], acc[mt][nt]);
    __syncthreads();
  }

#pragma unroll
  for (int nt = 0; nt < 4; ++nt) {
    int nCol = nBase + wn * 64 + nt * 16 + ln;
    float bval = bo[nCol];
#pragma unroll
    for (int mt = 0; mt < 2; ++mt) {
#pragma unroll
      for (int r = 0; r < 8; ++r) {
        int g = mBase + wm * 32 + mt * 16 + half * 8 + r;
        out[(size_t)g * DMODEL + nCol] = acc[mt][nt][r] + bval;
      }
    }
  }
}

// ---------------------------------------------------------------------------
extern "C" void kernel_launch(void* const* d_in, const int* in_sizes, int n_in,
                              void* d_out, int out_size, void* d_ws,
                              size_t ws_size, hipStream_t stream) {
  const float* query = (const float*)d_in[0];
  const float* keyin = (const float*)d_in[1];
  const float* value = (const float*)d_in[2];
  const float* Wq = (const float*)d_in[3];
  const float* bq = (const float*)d_in[4];
  const float* Wk = (const float*)d_in[5];
  const float* bk = (const float*)d_in[6];
  const float* Wv = (const float*)d_in[7];
  const float* bv = (const float*)d_in[8];
  const float* Wo = (const float*)d_in[9];
  const float* bo = (const float*)d_in[10];

  // workspace: Qh | Kh | Vt | attn_out, each B*H*S*DH f16 = 16 MB (64 MB total)
  const size_t seg = (size_t)BATCH * NHEAD * SEQ * DHEAD * sizeof(_Float16);
  char* ws = (char*)d_ws;
  _Float16* Qh = (_Float16*)(ws + 0 * seg);
  _Float16* Kh = (_Float16*)(ws + 1 * seg);
  _Float16* Vt = (_Float16*)(ws + 2 * seg);
  _Float16* At = (_Float16*)(ws + 3 * seg);

  dim3 g1(MROWS / 128, DMODEL / 128, 3);
  qkv_proj_kernel<<<g1, 256, 0, stream>>>(query, keyin, value, Wq, Wk, Wv, bq,
                                          bk, bv, Qh, Kh, Vt);

  dim3 g2(SEQ / 64, NHEAD, BATCH);
  flash_attn_kernel<<<g2, 128, 0, stream>>>(Qh, Kh, Vt, At);

  dim3 g3(MROWS / 128, DMODEL / 128, 1);
  out_proj_kernel<<<g3, 256, 0, stream>>>(At, Wo, bo, (float*)d_out);
}
